// PatchCore_22883585753563
// MI455X (gfx1250) — compile-verified, matrix operand order checked by reading, not run on hardware
//
#include <hip/hip_runtime.h>
#include <math.h>

// ---------------- problem constants ----------------
#define D_FEAT   384
#define N_ROWS   4096           // 64*64 patches
#define M_LIB    65536          // coreset size
#define FMAP     64
#define IMG      512

// ---------------- GEMM tiling ----------------
#define ROWS_PER_WG 128         // 8 waves * 16 rows
#define COLS_ITER   64          // library columns staged per iteration
#define COL_SEGS    16          // split of M across grid.y
#define KSTEPS      12          // 384 / 32
#define LDA         392         // LDS row stride in bf16 elems (384 + 8 pad)

typedef __attribute__((ext_vector_type(16))) __bf16 v16bf;
typedef __attribute__((ext_vector_type(8)))  float  v8f;

// ---------------- workspace layout (bytes) ----------------
#define WS_B2      0u           // 65536 f
#define WS_A2      262144u      // 4096 f
#define WS_ROWMIN  278528u      // 4096 u64
#define WS_MINVAL  311296u      // 4096 f
#define WS_MINIDX  327680u      // 4096 i
#define WS_SMAX    344064u      // 1 u64 (padded)
#define WS_W2      344320u      // 65536 f
#define WS_T2      606464u      // 65536 f
#define WS_REZ     868608u      // 512*512 f
#define WS_TMP     1917184u     // 512*512 f
#define WS_LIBBF   2965760u     // 65536*384 bf16
#define WS_PATCHBF 53297408u    // 4096*384 bf16
#define WS_NEED    56443136ull

// ---------------- helpers ----------------
static __device__ __forceinline__ unsigned long long shfl_xor_u64(unsigned long long v, int m) {
    unsigned int lo = (unsigned int)(v & 0xffffffffull);
    unsigned int hi = (unsigned int)(v >> 32);
    lo = (unsigned int)__shfl_xor((int)lo, m, 32);
    hi = (unsigned int)__shfl_xor((int)hi, m, 32);
    return (((unsigned long long)hi) << 32) | (unsigned long long)lo;
}

static __device__ __forceinline__ float wave_sum32(float s) {
#pragma unroll
    for (int m = 16; m >= 1; m >>= 1) s += __shfl_xor(s, m, 32);
    return s;
}

static __device__ __forceinline__ unsigned long long pack_di(float d, unsigned idx) {
    // d >= 0  =>  float bit pattern ordering == value ordering
    return (((unsigned long long)__float_as_uint(d)) << 32) | (unsigned long long)idx;
}

// gfx1250 async copy global -> LDS (16B), tracked by ASYNCcnt
static __device__ __forceinline__ void async_cp16(unsigned lds_off, unsigned goff,
                                                  unsigned long long gbase) {
    asm volatile("global_load_async_to_lds_b128 %0, %1, %2"
                 :: "v"(lds_off), "v"(goff), "s"(gbase)
                 : "memory");
}
static __device__ __forceinline__ void wait_async0() {
    asm volatile("s_wait_asynccnt 0x0" ::: "memory");
}

// gaussian(4, sigma=4) normalized 9-tap weights
__device__ __constant__ float GW[9] = {
    0.0816747f, 0.1016458f, 0.1188358f, 0.1305157f, 0.1346587f,
    0.1305157f, 0.1188358f, 0.1016458f, 0.0816747f
};

// ---------------- K0: squared norms of lib rows and patch rows (1 wave / row) ----
__global__ __launch_bounds__(256) void norms_kernel(
    const float* __restrict__ patch, const float* __restrict__ lib,
    float* __restrict__ a2, float* __restrict__ b2)
{
    int gw   = (blockIdx.x * 256 + threadIdx.x) >> 5;
    int lane = threadIdx.x & 31;
    const float* src;
    float* dst;
    if (gw < M_LIB) { src = lib + (size_t)gw * D_FEAT;            dst = b2 + gw; }
    else            { int r = gw - M_LIB; src = patch + (size_t)r * D_FEAT; dst = a2 + r; }
    float s = 0.f;
    for (int c = lane; c < D_FEAT; c += 32) { float v = src[c]; s += v * v; }
    s = wave_sum32(s);
    if (lane == 0) *dst = s;
}

// ---------------- K1: init scratch ----------------
__global__ void init_kernel(unsigned long long* rowmin, unsigned long long* smax)
{
    int i = blockIdx.x * blockDim.x + threadIdx.x;
    if (i < N_ROWS) rowmin[i] = ~0ull;
    if (i == 0) smax[0] = 0ull;
}

// ---------------- K2a: one-time fp32 -> bf16 conversion of both matrices ------
__global__ __launch_bounds__(256) void convert_bf16_kernel(
    const float* __restrict__ lib, const float* __restrict__ patch,
    __bf16* __restrict__ libbf, __bf16* __restrict__ patchbf)
{
    size_t i = (size_t)blockIdx.x * 256 + threadIdx.x;   // float4 index
    const float* src; __bf16* dst; size_t base;
    if (i < (size_t)M_LIB * (D_FEAT / 4)) { src = lib;   dst = libbf;   base = i; }
    else { src = patch; dst = patchbf; base = i - (size_t)M_LIB * (D_FEAT / 4); }
    float4 v = ((const float4*)src)[base];
    __bf16* o = dst + base * 4;
    o[0] = (__bf16)v.x; o[1] = (__bf16)v.y; o[2] = (__bf16)v.z; o[3] = (__bf16)v.w;
}

// issue async copies for one 64-row bf16 library tile into LDS (padded rows)
static __device__ __forceinline__ void issue_tileB(const __bf16* Bbf, int colBase,
                                                   unsigned ldsBase, int tid)
{
    unsigned long long gbase = (unsigned long long)(size_t)(Bbf + (size_t)colBase * D_FEAT);
    for (int i = tid; i < COLS_ITER * (D_FEAT / 8); i += 256) {   // 16B chunks
        int r = i / (D_FEAT / 8), c = i % (D_FEAT / 8);
        async_cp16(ldsBase + (unsigned)((r * LDA + c * 8) * 2),
                   (unsigned)((r * D_FEAT + c * 8) * 2), gbase);
    }
}

// ---------------- K2b: fused cdist + row min/argmin, async-LDS double buffer ---
__global__ __launch_bounds__(256) void dist_min_async(
    const __bf16* __restrict__ Abf, const __bf16* __restrict__ Bbf,
    const float* __restrict__ a2,   const float* __restrict__ b2,
    unsigned long long* __restrict__ rowmin, int segCols)
{
    extern __shared__ __align__(16) char smem[];
    __bf16* Abuf  = (__bf16*)smem;                       // ROWS_PER_WG x LDA
    __bf16* Bbuf0 = Abuf  + ROWS_PER_WG * LDA;           // COLS_ITER  x LDA
    __bf16* Bbuf1 = Bbuf0 + COLS_ITER  * LDA;            // COLS_ITER  x LDA

    const int tid  = threadIdx.x;
    const int lane = tid & 31;
    const int wave = tid >> 5;                           // 0..7
    const int mrow = lane & 15;
    const int lh8  = (lane >> 4) * 8;

    const int rowBase  = blockIdx.x * ROWS_PER_WG;
    const int colBase0 = blockIdx.y * segCols;
    const int nIter    = segCols / COLS_ITER;

    const unsigned aLds  = (unsigned)(size_t)Abuf;
    const unsigned bLds0 = (unsigned)(size_t)Bbuf0;
    const unsigned bLds1 = (unsigned)(size_t)Bbuf1;

    // ---- async-stage A tile (bf16, already converted) ----
    {
        unsigned long long gbase =
            (unsigned long long)(size_t)(Abf + (size_t)rowBase * D_FEAT);
        for (int i = tid; i < ROWS_PER_WG * (D_FEAT / 8); i += 256) {
            int r = i / (D_FEAT / 8), c = i % (D_FEAT / 8);
            async_cp16(aLds + (unsigned)((r * LDA + c * 8) * 2),
                       (unsigned)((r * D_FEAT + c * 8) * 2), gbase);
        }
    }
    // ---- async-stage B chunk 0 ----
    issue_tileB(Bbf, colBase0, bLds0, tid);
    wait_async0();
    __syncthreads();

    // ---- persistent A fragments: 12 k-steps x 8 VGPRs ----
    v16bf afrag[KSTEPS];
#pragma unroll
    for (int k = 0; k < KSTEPS; ++k) {
        const __bf16* s = &Abuf[(wave * 16 + mrow) * LDA + k * 32 + lh8];
        v16bf t;
#pragma unroll
        for (int e = 0; e < 8; ++e) { t[e] = s[e]; t[e + 8] = s[e + 16]; }
        afrag[k] = t;
    }

    float a2v[8];
#pragma unroll
    for (int r = 0; r < 8; ++r) a2v[r] = a2[rowBase + wave * 16 + r + lh8];

    unsigned long long runpk[8];
#pragma unroll
    for (int r = 0; r < 8; ++r) runpk[r] = ~0ull;

    for (int it = 0; it < nIter; ++it) {
        const int colBase = colBase0 + it * COLS_ITER;
        // overlap: DMA the next chunk into the other LDS buffer
        if (it + 1 < nIter)
            issue_tileB(Bbf, colBase + COLS_ITER, ((it + 1) & 1) ? bLds1 : bLds0, tid);
        if (it + 2 < nIter)   // keep chunk it+2 warming toward L2 (global_prefetch_b8)
            __builtin_prefetch(Bbf + (size_t)(colBase + 2 * COLS_ITER) * D_FEAT + tid * 96, 0, 1);

        const __bf16* Bcur = (it & 1) ? Bbuf1 : Bbuf0;
        float rb2v[COLS_ITER / 16];
#pragma unroll
        for (int ct = 0; ct < COLS_ITER / 16; ++ct)
            rb2v[ct] = b2[colBase + ct * 16 + mrow];

#pragma unroll
        for (int ct = 0; ct < COLS_ITER / 16; ++ct) {
            v8f acc = {0.f, 0.f, 0.f, 0.f, 0.f, 0.f, 0.f, 0.f};
            const __bf16* brow = &Bcur[(ct * 16 + mrow) * LDA + lh8];
            v16bf bcur;
#pragma unroll
            for (int e = 0; e < 8; ++e) { bcur[e] = brow[e]; bcur[e + 8] = brow[e + 16]; }
#pragma unroll
            for (int k = 0; k < KSTEPS; ++k) {
                v16bf bnext = bcur;                        // rotate so ds_load(k+1)
                if (k + 1 < KSTEPS) {                      // overlaps wmma(k)
                    const __bf16* bn = brow + (k + 1) * 32;
#pragma unroll
                    for (int e = 0; e < 8; ++e) { bnext[e] = bn[e]; bnext[e + 8] = bn[e + 16]; }
                }
                acc = __builtin_amdgcn_wmma_f32_16x16x32_bf16(
                        false, afrag[k], false, bcur, (short)0, acc, false, false);
                bcur = bnext;
            }
            const unsigned j = (unsigned)(colBase + ct * 16 + mrow);
#pragma unroll
            for (int r = 0; r < 8; ++r) {
                float d2 = fmaxf(a2v[r] + rb2v[ct] - 2.0f * acc[r], 0.0f);
                unsigned long long pk = pack_di(d2, j);
                runpk[r] = (pk < runpk[r]) ? pk : runpk[r];
            }
        }
        wait_async0();       // next chunk fully in LDS
        __syncthreads();     // visible to all waves; current buffer reusable
    }

#pragma unroll
    for (int r = 0; r < 8; ++r) {
        unsigned long long v = runpk[r];
#pragma unroll
        for (int m = 1; m <= 8; m <<= 1) {
            unsigned long long o = shfl_xor_u64(v, m);
            v = (o < v) ? o : v;
        }
        if (mrow == 0)
            atomicMin(&rowmin[rowBase + wave * 16 + r + lh8], v);
    }
}

// ---------------- K2-fallback: fp32 in-kernel staging (small workspace) -------
__global__ __launch_bounds__(256) void dist_min_fp32stage(
    const float* __restrict__ patch, const float* __restrict__ lib,
    const float* __restrict__ a2,    const float* __restrict__ b2,
    unsigned long long* __restrict__ rowmin, int segCols)
{
    extern __shared__ __align__(16) char smem[];
    __bf16* Abuf = (__bf16*)smem;
    __bf16* Bbuf = Abuf + ROWS_PER_WG * LDA;
    float*  B2s  = (float*)(Bbuf + COLS_ITER * LDA);

    const int tid  = threadIdx.x;
    const int lane = tid & 31;
    const int wave = tid >> 5;
    const int mrow = lane & 15;
    const int lh8  = (lane >> 4) * 8;

    const int rowBase  = blockIdx.x * ROWS_PER_WG;
    const int colBase0 = blockIdx.y * segCols;
    const int nIter    = segCols / COLS_ITER;

    for (int i = tid; i < ROWS_PER_WG * (D_FEAT / 4); i += 256) {
        int r = i / (D_FEAT / 4), c4 = i % (D_FEAT / 4);
        float4 v = ((const float4*)(patch + (size_t)(rowBase + r) * D_FEAT))[c4];
        __bf16* dst = &Abuf[r * LDA + c4 * 4];
        dst[0] = (__bf16)v.x; dst[1] = (__bf16)v.y;
        dst[2] = (__bf16)v.z; dst[3] = (__bf16)v.w;
    }
    __syncthreads();

    v16bf afrag[KSTEPS];
#pragma unroll
    for (int k = 0; k < KSTEPS; ++k) {
        const __bf16* s = &Abuf[(wave * 16 + mrow) * LDA + k * 32 + lh8];
        v16bf t;
#pragma unroll
        for (int e = 0; e < 8; ++e) { t[e] = s[e]; t[e + 8] = s[e + 16]; }
        afrag[k] = t;
    }

    float a2v[8];
#pragma unroll
    for (int r = 0; r < 8; ++r) a2v[r] = a2[rowBase + wave * 16 + r + lh8];

    unsigned long long runpk[8];
#pragma unroll
    for (int r = 0; r < 8; ++r) runpk[r] = ~0ull;

    for (int it = 0; it < nIter; ++it) {
        const int colBase = colBase0 + it * COLS_ITER;
        __syncthreads();
        for (int i = tid; i < COLS_ITER * (D_FEAT / 4); i += 256) {
            int r = i / (D_FEAT / 4), c4 = i % (D_FEAT / 4);
            float4 v = ((const float4*)(lib + (size_t)(colBase + r) * D_FEAT))[c4];
            __bf16* dst = &Bbuf[r * LDA + c4 * 4];
            dst[0] = (__bf16)v.x; dst[1] = (__bf16)v.y;
            dst[2] = (__bf16)v.z; dst[3] = (__bf16)v.w;
        }
        if (tid < COLS_ITER) B2s[tid] = b2[colBase + tid];
        if (it + 1 < nIter)
            __builtin_prefetch(lib + (size_t)(colBase + COLS_ITER) * D_FEAT + tid * 96, 0, 1);
        __syncthreads();

#pragma unroll
        for (int ct = 0; ct < COLS_ITER / 16; ++ct) {
            v8f acc = {0.f, 0.f, 0.f, 0.f, 0.f, 0.f, 0.f, 0.f};
            const __bf16* brow = &Bbuf[(ct * 16 + mrow) * LDA + lh8];
            v16bf bcur;
#pragma unroll
            for (int e = 0; e < 8; ++e) { bcur[e] = brow[e]; bcur[e + 8] = brow[e + 16]; }
#pragma unroll
            for (int k = 0; k < KSTEPS; ++k) {
                v16bf bnext = bcur;
                if (k + 1 < KSTEPS) {
                    const __bf16* bn = brow + (k + 1) * 32;
#pragma unroll
                    for (int e = 0; e < 8; ++e) { bnext[e] = bn[e]; bnext[e + 8] = bn[e + 16]; }
                }
                acc = __builtin_amdgcn_wmma_f32_16x16x32_bf16(
                        false, afrag[k], false, bcur, (short)0, acc, false, false);
                bcur = bnext;
            }
            const float    rb2 = B2s[ct * 16 + mrow];
            const unsigned j   = (unsigned)(colBase + ct * 16 + mrow);
#pragma unroll
            for (int r = 0; r < 8; ++r) {
                float d2 = fmaxf(a2v[r] + rb2 - 2.0f * acc[r], 0.0f);
                unsigned long long pk = pack_di(d2, j);
                runpk[r] = (pk < runpk[r]) ? pk : runpk[r];
            }
        }
    }

#pragma unroll
    for (int r = 0; r < 8; ++r) {
        unsigned long long v = runpk[r];
#pragma unroll
        for (int m = 1; m <= 8; m <<= 1) {
            unsigned long long o = shfl_xor_u64(v, m);
            v = (o < v) ? o : v;
        }
        if (mrow == 0)
            atomicMin(&rowmin[rowBase + wave * 16 + r + lh8], v);
    }
}

// ---------------- K3: unpack row mins, sqrt, global argmax ----------------
__global__ void finalize_rows(const unsigned long long* __restrict__ rowmin,
                              float* __restrict__ minval, int* __restrict__ minidx,
                              unsigned long long* __restrict__ smax)
{
    int i = blockIdx.x * blockDim.x + threadIdx.x;
    if (i >= N_ROWS) return;
    unsigned long long p = rowmin[i];
    float mv = sqrtf(__uint_as_float((unsigned)(p >> 32)));
    minval[i] = mv;
    minidx[i] = (int)(unsigned)(p & 0xffffffffull);
    atomicMax(smax, pack_di(mv, (unsigned)i));
}

// ---------------- K4: distances from m_star / m_test to the whole library ----
__global__ __launch_bounds__(256) void wt_kernel(
    const float* __restrict__ patch, const float* __restrict__ lib,
    const int* __restrict__ minidx, const unsigned long long* __restrict__ smax,
    float* __restrict__ w2, float* __restrict__ t2)
{
    int j    = (blockIdx.x * 256 + threadIdx.x) >> 5;
    int lane = threadIdx.x & 31;
    if (j >= M_LIB) return;
    int s_idx = (int)(unsigned)(smax[0] & 0xffffffffull);
    const float* mtest = patch + (size_t)s_idx * D_FEAT;
    const float* mstar = lib + (size_t)minidx[s_idx] * D_FEAT;
    const float* bj    = lib + (size_t)j * D_FEAT;
    float sw = 0.f, st = 0.f;
    for (int c = lane; c < D_FEAT; c += 32) {
        float b  = bj[c];
        float dw = mstar[c] - b; sw += dw * dw;
        float dt = mtest[c] - b; st += dt * dt;
    }
    sw = wave_sum32(sw);
    st = wave_sum32(st);
    if (lane == 0) { w2[j] = sw; t2[j] = st; }
}

// ---------------- K5: top-3 nearest of m_star + reweighted score s ----------
__global__ __launch_bounds__(256) void topk_kernel(
    const float* __restrict__ w2, const float* __restrict__ t2,
    const unsigned long long* __restrict__ smax, float* __restrict__ out_s)
{
    __shared__ unsigned long long red[256];
    int tid = threadIdx.x;
    int found[3];
    for (int k = 0; k < 3; ++k) {
        unsigned long long loc = ~0ull;
        for (int j = tid; j < M_LIB; j += 256) {
            bool skip = false;
            for (int q = 0; q < k; ++q) if (found[q] == j) skip = true;
            if (skip) continue;
            unsigned long long pk = pack_di(fmaxf(w2[j], 0.f), (unsigned)j);
            loc = (pk < loc) ? pk : loc;
        }
        red[tid] = loc;
        __syncthreads();
        for (int s = 128; s > 0; s >>= 1) {
            if (tid < s && red[tid + s] < red[tid]) red[tid] = red[tid + s];
            __syncthreads();
        }
        found[k] = (int)(unsigned)(red[0] & 0xffffffffull);
        __syncthreads();
    }
    if (tid == 0) {
        float s_star = __uint_as_float((unsigned)(smax[0] >> 32));
        float invD   = 1.0f / sqrtf((float)D_FEAT);
        float k1 = sqrtf(fmaxf(t2[found[1]], 0.f));
        float k2 = sqrtf(fmaxf(t2[found[2]], 0.f));
        float w  = 1.0f - expf(s_star * invD) / (expf(k1 * invD) + expf(k2 * invD));
        out_s[0] = w * s_star;
    }
}

// ---------------- K6: 64x64 -> 512x512 bilinear (half-pixel centers) --------
__global__ __launch_bounds__(256) void resize_kernel(
    const float* __restrict__ mv, float* __restrict__ out)
{
    int i = blockIdx.x * blockDim.x + threadIdx.x;
    if (i >= IMG * IMG) return;
    int oy = i >> 9, ox = i & (IMG - 1);
    float sy = (oy + 0.5f) * ((float)FMAP / IMG) - 0.5f;
    float sx = (ox + 0.5f) * ((float)FMAP / IMG) - 0.5f;
    int y0 = (int)floorf(sy), x0 = (int)floorf(sx);
    float fy = sy - y0, fx = sx - x0;
    int y0c = min(max(y0, 0), FMAP - 1), y1c = min(max(y0 + 1, 0), FMAP - 1);
    int x0c = min(max(x0, 0), FMAP - 1), x1c = min(max(x0 + 1, 0), FMAP - 1);
    float v00 = mv[y0c * FMAP + x0c], v01 = mv[y0c * FMAP + x1c];
    float v10 = mv[y1c * FMAP + x0c], v11 = mv[y1c * FMAP + x1c];
    out[i] = (1.f - fy) * ((1.f - fx) * v00 + fx * v01)
           +        fy  * ((1.f - fx) * v10 + fx * v11);
}

// ---------------- K7/K8: separable 9-tap gaussian, zero ('SAME') padding ----
__global__ __launch_bounds__(256) void blurH_kernel(
    const float* __restrict__ in, float* __restrict__ out)
{
    int i = blockIdx.x * blockDim.x + threadIdx.x;
    if (i >= IMG * IMG) return;
    int y = i >> 9, x = i & (IMG - 1);
    float a = 0.f;
#pragma unroll
    for (int k = -4; k <= 4; ++k) {
        int xx = x + k;
        if (xx >= 0 && xx < IMG) a += GW[k + 4] * in[y * IMG + xx];
    }
    out[i] = a;
}

__global__ __launch_bounds__(256) void blurV_kernel(
    const float* __restrict__ in, float* __restrict__ out)
{
    int i = blockIdx.x * blockDim.x + threadIdx.x;
    if (i >= IMG * IMG) return;
    int y = i >> 9, x = i & (IMG - 1);
    float a = 0.f;
#pragma unroll
    for (int k = -4; k <= 4; ++k) {
        int yy = y + k;
        if (yy >= 0 && yy < IMG) a += GW[k + 4] * in[yy * IMG + x];
    }
    out[i] = a;
}

// ---------------- host launch ----------------
extern "C" void kernel_launch(void* const* d_in, const int* in_sizes, int n_in,
                              void* d_out, int out_size, void* d_ws, size_t ws_size,
                              hipStream_t stream) {
    const float* patch = (const float*)d_in[0];   // [4096, 384]
    const float* lib   = (const float*)d_in[1];   // [65536, 384]
    float* out = (float*)d_out;                   // [1 + 512*512]
    char*  ws  = (char*)d_ws;

    float*              b2      = (float*)(ws + WS_B2);
    float*              a2      = (float*)(ws + WS_A2);
    unsigned long long* rowmin  = (unsigned long long*)(ws + WS_ROWMIN);
    float*              minval  = (float*)(ws + WS_MINVAL);
    int*                minidx  = (int*)(ws + WS_MINIDX);
    unsigned long long* smax    = (unsigned long long*)(ws + WS_SMAX);
    float*              w2      = (float*)(ws + WS_W2);
    float*              t2      = (float*)(ws + WS_T2);
    float*              rez     = (float*)(ws + WS_REZ);
    float*              tmp     = (float*)(ws + WS_TMP);
    __bf16*             libbf   = (__bf16*)(ws + WS_LIBBF);
    __bf16*             patchbf = (__bf16*)(ws + WS_PATCHBF);

    norms_kernel<<<(M_LIB + N_ROWS) / 8, 256, 0, stream>>>(patch, lib, a2, b2);
    init_kernel<<<(N_ROWS + 255) / 256, 256, 0, stream>>>(rowmin, smax);

    if (ws_size >= WS_NEED) {
        // preferred path: bf16 pre-convert + async global->LDS double buffering
        convert_bf16_kernel<<<(M_LIB + N_ROWS) * (D_FEAT / 4) / 256, 256, 0, stream>>>(
            lib, patch, libbf, patchbf);
        size_t shbytes = (size_t)(ROWS_PER_WG * LDA + 2 * COLS_ITER * LDA) * sizeof(__bf16);
        dist_min_async<<<dim3(N_ROWS / ROWS_PER_WG, COL_SEGS), 256, shbytes, stream>>>(
            patchbf, libbf, a2, b2, rowmin, M_LIB / COL_SEGS);
    } else {
        size_t shbytes = (size_t)(ROWS_PER_WG * LDA + COLS_ITER * LDA) * sizeof(__bf16)
                       + COLS_ITER * sizeof(float);
        dist_min_fp32stage<<<dim3(N_ROWS / ROWS_PER_WG, COL_SEGS), 256, shbytes, stream>>>(
            patch, lib, a2, b2, rowmin, M_LIB / COL_SEGS);
    }

    finalize_rows<<<(N_ROWS + 255) / 256, 256, 0, stream>>>(rowmin, minval, minidx, smax);
    wt_kernel<<<M_LIB / 8, 256, 0, stream>>>(patch, lib, minidx, smax, w2, t2);
    topk_kernel<<<1, 256, 0, stream>>>(w2, t2, smax, out);

    resize_kernel<<<(IMG * IMG) / 256, 256, 0, stream>>>(minval, rez);
    blurH_kernel<<<(IMG * IMG) / 256, 256, 0, stream>>>(rez, tmp);
    blurV_kernel<<<(IMG * IMG) / 256, 256, 0, stream>>>(tmp, out + 1);
}
